// HybridBlock_80530636800511
// MI455X (gfx1250) — compile-verified, matrix operand order checked by reading, not run on hardware
//
#include <hip/hip_runtime.h>
#include <hip/hip_bf16.h>
#include <math.h>

// ---------------------------------------------------------------------------
// MI455X hybrid block: minGRU + sliding-window attention + GLU FFN
// All large GEMMs run through v_wmma_f32_16x16x32_bf16 (wave32, 16x16x32).
// Activation tiles (already bf16) are staged LDS-direct with
// GLOBAL_LOAD_ASYNC_TO_LDS_B128 (ASYNCcnt) and double-buffered so WMMA
// overlaps the async copies. Weight tiles are converted f32->bf16 on the fly
// (all weights fit in the 192MB L2, so fp32 re-reads are cheap).
// ---------------------------------------------------------------------------

typedef __attribute__((ext_vector_type(16))) __bf16 v16bf;
typedef __attribute__((ext_vector_type(8)))  float  v8f;

static __device__ inline unsigned short f2bf(float f) {
    union { float f; unsigned int u; } v; v.f = f;
    unsigned int r = v.u + 0x7FFFu + ((v.u >> 16) & 1u);   // round-to-nearest-even
    return (unsigned short)(r >> 16);
}

static __device__ inline float sigmoidf_(float x) { return 1.0f / (1.0f + expf(-x)); }

// Async byte-copy global -> LDS, 16B per lane, tracked by ASYNCcnt.
// lds_off is the 32-bit LDS byte address (low 32 bits of a generic pointer
// into shared memory, per the CDNA5 aperture mapping).
static __device__ inline void async_load_b128(unsigned lds_off, const void* gptr) {
    asm volatile("global_load_async_to_lds_b128 %0, %1, off"
                 :: "v"(lds_off), "v"(gptr) : "memory");
}

static __device__ inline void wait_async0() {
#if defined(__has_builtin)
#if __has_builtin(__builtin_amdgcn_s_wait_asynccnt)
    __builtin_amdgcn_s_wait_asynccnt(0);
#else
    asm volatile("s_wait_asynccnt 0x0" ::: "memory");
#endif
#else
    asm volatile("s_wait_asynccnt 0x0" ::: "memory");
#endif
}

// Load one 16x32 bf16 WMMA fragment (A- or B-style) from a 32-element K-slice.
// rowU32 points at the 16 dwords of this row's K-slice in LDS.
// Layout per CDNA5 ISA 7.12.2 (16-bit A 16x32): lanes 0-15 hold K 0..7 /
// 16..23, lanes 16-31 hold K 8..15 / 24..31; VGPR j pairs K {2j,2j+1}.
static __device__ inline v16bf load_frag(const unsigned int* rowU32, int lane) {
    union { unsigned int u[8]; v16bf v; } f;
    int base = ((lane >> 4) & 1) * 4;      // K-offset 8 for upper half-wave
#pragma unroll
    for (int j = 0; j < 4; ++j) f.u[j] = rowU32[base + j];        // K 0..7 (+half)
#pragma unroll
    for (int j = 0; j < 4; ++j) f.u[4 + j] = rowU32[base + j + 8]; // K 16..23 (+half)
    return f.v;
}

static __device__ inline v8f v8f_zero() {
    v8f z = {0.f, 0.f, 0.f, 0.f, 0.f, 0.f, 0.f, 0.f};
    return z;
}

// ---------------------------------------------------------------------------
// RMSNorm: one block per row, output bf16
// ---------------------------------------------------------------------------
__global__ __launch_bounds__(256) void rmsnorm_bf16_kernel(
    const float* __restrict__ x, const float* __restrict__ w,
    unsigned short* __restrict__ out, int D)
{
    __shared__ float red[256];
    size_t row = blockIdx.x;
    const float* xr = x + row * (size_t)D;
    float s = 0.f;
    for (int i = threadIdx.x; i < D; i += 256) { float v = xr[i]; s += v * v; }
    red[threadIdx.x] = s;
    __syncthreads();
    for (int st = 128; st > 0; st >>= 1) {
        if (threadIdx.x < st) red[threadIdx.x] += red[threadIdx.x + st];
        __syncthreads();
    }
    float inv = rsqrtf(red[0] / (float)D + 1e-6f);
    unsigned short* orow = out + row * (size_t)D;
    for (int i = threadIdx.x; i < D; i += 256)
        orow[i] = f2bf(xr[i] * inv * w[i]);
}

// ---------------------------------------------------------------------------
// WMMA GEMM: C[M,N] f32 = A_bf16[M,K] @ W_f32[N,K]^T (+bias[N]) (+resid[M,N])
// Block tile 128x128x32, 8 waves, each wave 64x32 (4x2 16x16 tiles).
// Double-buffered LDS; A tile staged with async global->LDS b128 copies.
// ---------------------------------------------------------------------------
#define BM 128
#define BN 128
#define BK 32

__global__ __launch_bounds__(256) void gemm_bf16_wmma(
    const unsigned short* __restrict__ A, const float* __restrict__ W,
    const float* __restrict__ bias, const float* __restrict__ resid,
    float* __restrict__ C, int M, int N, int K)
{
    __shared__ __align__(16) unsigned short As[2][BM][BK];   // 2 x 8KB
    __shared__ __align__(16) unsigned short Bs[2][BN][BK];   // 2 x 8KB

    int tid  = threadIdx.x;
    int lane = tid & 31;
    int wid  = tid >> 5;
    int wm   = wid >> 2;        // 0..1 -> 64-row half
    int wn   = wid & 3;         // 0..3 -> 32-col quarter
    int bn0  = blockIdx.x * BN;
    int bm0  = blockIdx.y * BM;

    v8f acc[4][2];
#pragma unroll
    for (int mt = 0; mt < 4; ++mt)
#pragma unroll
        for (int nt = 0; nt < 2; ++nt) acc[mt][nt] = v8f_zero();

    const char* Abytes = (const char*)A;

    // Stage the A tile (bf16 byte copy) into LDS buffer p via async b128 ops:
    // 128 rows x 64B = 512 x 16B chunks, 2 per thread.
    auto stageA = [&](int k0, int p) {
#pragma unroll
        for (int j = 0; j < 2; ++j) {
            int cid = tid + 256 * j;          // 0..511
            int row = cid >> 2, q = cid & 3;
            const char* g = Abytes + ((size_t)(bm0 + row) * K + k0) * 2 + q * 16;
            unsigned l = (unsigned)(size_t)&As[p][row][q * 8];
            async_load_b128(l, g);
        }
    };
    // Stage the W tile with f32 -> bf16 conversion (16 values per thread).
    auto stageB = [&](int k0, int p) {
        int row = tid >> 1, half = tid & 1;
        const float* src = W + (size_t)(bn0 + row) * K + k0 + half * 16;
        unsigned short* dst = &Bs[p][row][half * 16];
#pragma unroll
        for (int j = 0; j < 16; ++j) dst[j] = f2bf(src[j]);
    };

    stageA(0, 0);
    stageB(0, 0);
    wait_async0();
    __syncthreads();

    int nk = K / BK;
    for (int it = 0; it < nk; ++it) {
        int p = it & 1;
        if (it + 1 < nk) {                    // software pipeline: prefetch next
            stageA((it + 1) * BK, 1 - p);
            stageB((it + 1) * BK, 1 - p);
            __builtin_prefetch(W + (size_t)(bn0 + (tid >> 1)) * K + (it + 1) * BK, 0, 0);
        }

        v16bf af[4], bfr[2];
#pragma unroll
        for (int mt = 0; mt < 4; ++mt)
            af[mt] = load_frag(
                (const unsigned int*)&As[p][wm * 64 + mt * 16 + (lane & 15)][0], lane);
#pragma unroll
        for (int nt = 0; nt < 2; ++nt)
            bfr[nt] = load_frag(
                (const unsigned int*)&Bs[p][wn * 32 + nt * 16 + (lane & 15)][0], lane);
#pragma unroll
        for (int mt = 0; mt < 4; ++mt)
#pragma unroll
            for (int nt = 0; nt < 2; ++nt)
                acc[mt][nt] = __builtin_amdgcn_wmma_f32_16x16x32_bf16(
                    false, af[mt], false, bfr[nt], (short)0, acc[mt][nt],
                    false, false);

        wait_async0();                        // async writes for tile it+1 done
        __syncthreads();                      // visible to all waves
    }

    // Epilogue: C-tile layout (7.12.2): lane -> N = lane&15, VGPR j -> M = j + 8*(lane>>4)
#pragma unroll
    for (int mt = 0; mt < 4; ++mt) {
#pragma unroll
        for (int nt = 0; nt < 2; ++nt) {
            int col = bn0 + wn * 32 + nt * 16 + (lane & 15);
            float bv = bias ? bias[col] : 0.f;
#pragma unroll
            for (int jj = 0; jj < 8; ++jj) {
                int row = bm0 + wm * 64 + mt * 16 + ((lane >> 4) * 8) + jj;
                size_t idx = (size_t)row * N + col;
                float val = acc[mt][nt][jj] + bv;
                if (resid) val += resid[idx];
                C[idx] = val;
            }
        }
    }
}

// ---------------------------------------------------------------------------
// minGRU elementwise: xs = sigmoid(g)*tanh(v) (into g), a = .001+.998*sig(d) (into d)
// ---------------------------------------------------------------------------
__global__ __launch_bounds__(256) void gru_gate_kernel(
    float* __restrict__ g, const float* __restrict__ v, float* __restrict__ d,
    size_t n)
{
    size_t i = (size_t)blockIdx.x * 256 + threadIdx.x;
    if (i >= n) return;
    float gg = g[i], vv = v[i], dd = d[i];
    g[i] = sigmoidf_(gg) * tanhf(vv);
    d[i] = 0.001f + 0.998f * sigmoidf_(dd);
}

// ---------------------------------------------------------------------------
// 3-pass chunked linear scan  h_t = a_t h_{t-1} + x_t   (16 chunks of 128)
// ---------------------------------------------------------------------------
__global__ __launch_bounds__(256) void scan_pass1(
    const float* __restrict__ a, const float* __restrict__ xs,
    float* __restrict__ CA, float* __restrict__ CH,
    int S, int D, int nch, int clen)
{
    int idx = blockIdx.x * 256 + threadIdx.x;   // (b, ch, d)
    int d  = idx % D;
    int r  = idx / D;
    int ch = r % nch;
    int b  = r / nch;
    float ap = 1.f, hp = 0.f;
    for (int t = 0; t < clen; ++t) {
        size_t off = ((size_t)b * S + (size_t)ch * clen + t) * D + d;
        float av = a[off];
        ap = ap * av;
        hp = av * hp + xs[off];
    }
    size_t ci = ((size_t)b * nch + ch) * D + d;
    CA[ci] = ap;
    CH[ci] = hp;
}

__global__ __launch_bounds__(256) void scan_pass2(
    float* __restrict__ CA, const float* __restrict__ CH, int D, int nch)
{
    int idx = blockIdx.x * 256 + threadIdx.x;   // (b, d)
    int d = idx % D;
    int b = idx / D;
    float ex = 0.f;
    for (int ch = 0; ch < nch; ++ch) {
        size_t ci = ((size_t)b * nch + ch) * D + d;
        float ne = CA[ci] * ex + CH[ci];
        CA[ci] = ex;                 // exclusive carry into chunk ch
        ex = ne;
    }
}

__global__ __launch_bounds__(256) void scan_pass3(
    const float* __restrict__ x, const float* __restrict__ a,
    const float* __restrict__ xs, const float* __restrict__ CA,
    float* __restrict__ x1, int S, int D, int nch, int clen)
{
    int idx = blockIdx.x * 256 + threadIdx.x;
    int d  = idx % D;
    int r  = idx / D;
    int ch = r % nch;
    int b  = r / nch;
    float h = CA[((size_t)b * nch + ch) * D + d];
    for (int t = 0; t < clen; ++t) {
        size_t off = ((size_t)b * S + (size_t)ch * clen + t) * D + d;
        h = a[off] * h + xs[off];
        x1[off] = x[off] + h;        // fused residual
    }
}

// ---------------------------------------------------------------------------
// Sliding-window attention, one block per (b, h, chunk, 32-query slab).
// qkv: [B,S,3,H,64] f32.  out: [B,S,H*64] bf16.  60KB static LDS.
// ---------------------------------------------------------------------------
__global__ __launch_bounds__(256) void attn_kernel(
    const float* __restrict__ qkv, unsigned short* __restrict__ out,
    int Bb, int S)
{
    const int H = 16, DH = 64, WIN = 128, TW = 256, QR = 32;
    __shared__ __align__(16) unsigned short Qs[QR][DH];   //  4KB
    __shared__ __align__(16) unsigned short KVs[64][64];  //  8KB (K tile; then V^T tile)
    __shared__ float          Sc[QR][TW];                 // 32KB
    __shared__ __align__(16) unsigned short Ps[QR][TW];   // 16KB

    int tid = threadIdx.x, lane = tid & 31, wid = tid >> 5;
    int nC = S / WIN;
    int t  = blockIdx.x;
    int qb = t & 3;        t >>= 2;
    int c  = t % nC;       t /= nC;
    int h  = t % H;        t /= H;
    int b  = t;
    int j0 = qb * QR;      // first query row in chunk for this block
    (void)Bb;

    // stage Q slab (32x64), f32 -> bf16
#pragma unroll
    for (int j = 0; j < 8; ++j) {
        int i = tid * 8 + j;
        int row = i >> 6, e = i & 63;
        int s = c * WIN + j0 + row;
        Qs[row][e] = f2bf(qkv[((((size_t)b * S + s) * 3 + 0) * H + h) * DH + e]);
    }

    // ----- phase 1: scores over 4 key tiles of 64 -----
    int mt = wid >> 2, nt = wid & 3;          // this wave's 16x16 score tile
    for (int kt = 0; kt < 4; ++kt) {
#pragma unroll
        for (int j = 0; j < 16; ++j) {        // stage K tile (64x64)
            int i = tid * 16 + j;
            int row = i >> 6, e = i & 63;
            int n = kt * 64 + row;
            int sk = (c - 1) * WIN + n;       // key position (prev|cur)
            int skc = sk < 0 ? 0 : sk;        // masked anyway when c==0
            KVs[row][e] = f2bf(qkv[((((size_t)b * S + skc) * 3 + 1) * H + h) * DH + e]);
        }
        __syncthreads();

        v8f sc = v8f_zero();
#pragma unroll
        for (int ks = 0; ks < 2; ++ks) {      // K-dim 64 = 2 x 32
            v16bf af = load_frag(
                (const unsigned int*)&Qs[mt * 16 + (lane & 15)][ks * 32], lane);
            v16bf bf_ = load_frag(
                (const unsigned int*)&KVs[nt * 16 + (lane & 15)][ks * 32], lane);
            sc = __builtin_amdgcn_wmma_f32_16x16x32_bf16(
                false, af, false, bf_, (short)0, sc, false, false);
        }
        int col = nt * 16 + (lane & 15);
        int kidx = kt * 64 + col;             // key index in [prev|cur] window
#pragma unroll
        for (int jj = 0; jj < 8; ++jj) {
            int m = mt * 16 + ((lane >> 4) * 8) + jj;
            int jq = j0 + m;                  // query pos in chunk
            bool ok = (kidx <= WIN + jq) && (kidx >= jq + 1) &&
                      ((kidx >= WIN) || (c > 0));
            Sc[m][kidx] = ok ? sc[jj] * 0.125f : -3.0e38f;
        }
        __syncthreads();
    }

    // ----- softmax (one thread per query row) -----
    if (tid < QR) {
        float mx = -3.0e38f;
        for (int k = 0; k < TW; ++k) mx = fmaxf(mx, Sc[tid][k]);
        float sum = 0.f;
        for (int k = 0; k < TW; ++k) sum += expf(Sc[tid][k] - mx);
        float inv = 1.f / sum;
        for (int k = 0; k < TW; ++k)
            Ps[tid][k] = f2bf(expf(Sc[tid][k] - mx) * inv);
    }
    __syncthreads();

    // ----- phase 2: O = P(32x256) @ V(256x64) -----
    v8f oacc = v8f_zero();
    for (int kt = 0; kt < 4; ++kt) {
#pragma unroll
        for (int j = 0; j < 16; ++j) {        // stage V tile TRANSPOSED: KVs[dh][key]
            int i = tid * 16 + j;
            int row = i >> 6, e = i & 63;
            int n = kt * 64 + row;
            int sk = (c - 1) * WIN + n;
            int skc = sk < 0 ? 0 : sk;
            KVs[e][row] = f2bf(qkv[((((size_t)b * S + skc) * 3 + 2) * H + h) * DH + e]);
        }
        __syncthreads();
#pragma unroll
        for (int ks = 0; ks < 2; ++ks) {
            v16bf af = load_frag(
                (const unsigned int*)&Ps[mt * 16 + (lane & 15)][kt * 64 + ks * 32],
                lane);
            v16bf bf_ = load_frag(
                (const unsigned int*)&KVs[nt * 16 + (lane & 15)][ks * 32], lane);
            oacc = __builtin_amdgcn_wmma_f32_16x16x32_bf16(
                false, af, false, bf_, (short)0, oacc, false, false);
        }
        __syncthreads();
    }

    // write O as bf16 into [B,S,H*64]
    int col = nt * 16 + (lane & 15);
#pragma unroll
    for (int jj = 0; jj < 8; ++jj) {
        int m = mt * 16 + ((lane >> 4) * 8) + jj;
        int s = c * WIN + j0 + m;
        out[((size_t)b * S + s) * (H * DH) + h * DH + col] = f2bf(oacc[jj]);
    }
}

// ---------------------------------------------------------------------------
// GLU elementwise: hb = bf16( silu(g) * u )
// ---------------------------------------------------------------------------
__global__ __launch_bounds__(256) void glu_kernel(
    const float* __restrict__ g, const float* __restrict__ u,
    unsigned short* __restrict__ hb, size_t n)
{
    size_t i = (size_t)blockIdx.x * 256 + threadIdx.x;
    if (i >= n) return;
    float gv = g[i];
    hb[i] = f2bf((gv * sigmoidf_(gv)) * u[i]);
}

// ---------------------------------------------------------------------------
extern "C" void kernel_launch(void* const* d_in, const int* in_sizes, int n_in,
                              void* d_out, int out_size, void* d_ws, size_t ws_size,
                              hipStream_t stream)
{
    (void)in_sizes; (void)n_in; (void)out_size; (void)ws_size;
    const int B = 4, S = 2048, D = 1024, DFF = 4096, H = 16;
    const size_t BS = (size_t)B * S;

    const float* x        = (const float*)d_in[0];
    const float* w_mix    = (const float*)d_in[1];
    const float* Wg       = (const float*)d_in[2];
    const float* bg       = (const float*)d_in[3];
    const float* Wv       = (const float*)d_in[4];
    const float* bv       = (const float*)d_in[5];
    const float* Wd       = (const float*)d_in[6];
    const float* bd       = (const float*)d_in[7];
    const float* w_attn   = (const float*)d_in[8];
    const float* Wqkv     = (const float*)d_in[9];
    const float* Wo       = (const float*)d_in[10];
    const float* w_ffn    = (const float*)d_in[11];
    const float* Wgate    = (const float*)d_in[12];
    const float* Wup      = (const float*)d_in[13];
    const float* Wout     = (const float*)d_in[14];
    float* out = (float*)d_out;

    // workspace layout (144.5 MB)
    char* ws = (char*)d_ws;
    float* Gb = (float*)(ws);                          // 32MB (g / qkv / ffn-gate)
    float* Vb = (float*)(ws + ((size_t)32 << 20));     // 32MB (v / ffn-up)
    float* Db = (float*)(ws + ((size_t)64 << 20));     // 32MB (d/a)
    float* X1 = (float*)(ws + ((size_t)96 << 20));     // 32MB (x1; later hb bf16)
    unsigned short* XN = (unsigned short*)(ws + ((size_t)128 << 20)); // 16MB
    float* CA = (float*)(ws + ((size_t)144 << 20));    // 256KB chunk A-products
    float* CH = (float*)(ws + ((size_t)144 << 20) + (1 << 18)); // 256KB chunk H
    float* QKV = Gb;                                   // 96MB spans Gb..Db

    dim3 blk(256);
    const int nch = 16, clen = S / nch;

    // ---- minGRU branch ----
    rmsnorm_bf16_kernel<<<(int)BS, blk, 0, stream>>>(x, w_mix, XN, D);
    gemm_bf16_wmma<<<dim3(D / BN, BS / BM), blk, 0, stream>>>(XN, Wg, bg, nullptr, Gb, BS, D, D);
    gemm_bf16_wmma<<<dim3(D / BN, BS / BM), blk, 0, stream>>>(XN, Wv, bv, nullptr, Vb, BS, D, D);
    gemm_bf16_wmma<<<dim3(D / BN, BS / BM), blk, 0, stream>>>(XN, Wd, bd, nullptr, Db, BS, D, D);
    gru_gate_kernel<<<(int)(BS * D / 256), blk, 0, stream>>>(Gb, Vb, Db, BS * D);
    scan_pass1<<<B * D * nch / 256, blk, 0, stream>>>(Db, Gb, CA, CH, S, D, nch, clen);
    scan_pass2<<<B * D / 256, blk, 0, stream>>>(CA, CH, D, nch);
    scan_pass3<<<B * D * nch / 256, blk, 0, stream>>>(x, Db, Gb, CA, X1, S, D, nch, clen);

    // ---- sliding-window attention branch ----
    rmsnorm_bf16_kernel<<<(int)BS, blk, 0, stream>>>(X1, w_attn, XN, D);
    gemm_bf16_wmma<<<dim3(3 * D / BN, BS / BM), blk, 0, stream>>>(XN, Wqkv, nullptr, nullptr, QKV, BS, 3 * D, D);
    attn_kernel<<<B * H * (S / 128) * 4, blk, 0, stream>>>(QKV, XN, B, S);
    gemm_bf16_wmma<<<dim3(D / BN, BS / BM), blk, 0, stream>>>(XN, Wo, nullptr, X1, out, BS, D, D);

    // ---- GLU FFN branch (4 row-chunks of 2048 to bound workspace) ----
    rmsnorm_bf16_kernel<<<(int)BS, blk, 0, stream>>>(out, w_ffn, XN, D);
    unsigned short* HB = (unsigned short*)X1;
    for (int ck = 0; ck < 4; ++ck) {
        const int Mc = 2048;
        const unsigned short* An = XN + (size_t)ck * Mc * D;
        float* ro = out + (size_t)ck * Mc * D;
        gemm_bf16_wmma<<<dim3(DFF / BN, Mc / BM), blk, 0, stream>>>(An, Wgate, nullptr, nullptr, Gb, Mc, DFF, D);
        gemm_bf16_wmma<<<dim3(DFF / BN, Mc / BM), blk, 0, stream>>>(An, Wup, nullptr, nullptr, Vb, Mc, DFF, D);
        glu_kernel<<<(int)((size_t)Mc * DFF / 256), blk, 0, stream>>>(Gb, Vb, HB, (size_t)Mc * DFF);
        gemm_bf16_wmma<<<dim3(D / BN, Mc / BM), blk, 0, stream>>>(HB, Wout, nullptr, ro, ro, Mc, D, DFF);
    }
}
